// TPMoERouter_15427522527440
// MI455X (gfx1250) — matrix operand order repaired and checked by reading, not compiled, not run on hardware
//
#include <hip/hip_runtime.h>
#include <stdint.h>

// MoE router for MI455X (gfx1250):
//   logits = x[32768,4096] @ W[64,4096]^T  via bf16 hi/lo split WMMA (fp32-accurate),
//   fused softmax/top-2 epilogue.
// Data path: TDM (tensor_load_to_lds) double-buffered K-chunks into LDS with
// padded rows for bank-conflict-free reads. W is pre-split to bf16 hi/lo in a
// 1MB pre-pass (L2-resident) so the hot loop does zero W conversion VALU.

typedef __attribute__((ext_vector_type(16))) __bf16 v16bf;
typedef __attribute__((ext_vector_type(8)))  float  v8f;

#define T_TOKENS 32768
#define HIDDEN   4096
#define EXPERTS  64
#define KC       32            // K per chunk (one bf16 WMMA k-step)
#define M_BLOCK  128           // tokens per workgroup (8 waves x 16)
#define XSTRIDE  33            // padded LDS row stride for f32 x-tile (dwords)
#define XBUF     (M_BLOCK * XSTRIDE + 16)    // 4240 dwords
#define WBUF     (EXPERTS * XSTRIDE + 16)    // 2128 dwords (f32 W, fallback mode)
#define WHSTRIDE 17                          // bf16 W row stride in dwords (34 bf16)
#define WHBUF    (EXPERTS * WHSTRIDE + 16)   // 1104 dwords (bf16 W tile, pre-split mode)
#define SMEM_DW  (2 * XBUF + 4 * WHBUF)      // 12896 dwords = 51,584 B (max of both modes)

#if __has_builtin(__builtin_amdgcn_tensor_load_to_lds)
#define USE_TDM 1
#else
#define USE_TDM 0
#endif

#if USE_TDM
// Build a 2D TDM descriptor (D#) per CDNA5 ISA and issue it.
// dims/strides in elements. LDS pad: 1 dword after every (1<<pad_iv) 8-byte units.
// clang-23 6-arg form: (uint32x4 g0, int32x8 g1, int32x4, int32x4, int32x8, i32 cpol)
__device__ __forceinline__ void tdm_issue_2d(unsigned lds_byte_off, const void* gptr,
                                             unsigned tensor_d0, unsigned tensor_d1,
                                             unsigned long long stride0,
                                             unsigned tile_d0, unsigned tile_d1,
                                             unsigned ds_code, unsigned pad_iv) {
  typedef unsigned int v4u __attribute__((ext_vector_type(4)));
  typedef int v8i_t __attribute__((ext_vector_type(8)));
  typedef int v4i_t __attribute__((ext_vector_type(4)));
  unsigned long long ga = (unsigned long long)(uintptr_t)gptr;
  v4u g0;
  g0[0] = 1u;                                   // count=1, user descriptor
  g0[1] = lds_byte_off;                         // lds_addr
  g0[2] = (unsigned)(ga & 0xffffffffu);         // global_addr[31:0]
  g0[3] = (unsigned)((ga >> 32) & 0x1ffffffu)   // global_addr[56:32]
        | (2u << 30);                           // type = 2 ("image")
  unsigned w0 = (ds_code << 16)                 // data_size: 1=2B, 2=4B
              | (1u << 20)                      // pad_enable
              | (pad_iv << 22)                  // pad_interval
              | (0u << 25);                     // pad_amount: 1 dword
  unsigned w1 = (tensor_d0 & 0xffffu) << 16;                       // dim0[15:0] @63:48
  unsigned w2 = (tensor_d0 >> 16) | ((tensor_d1 & 0xffffu) << 16); // dim0 hi, dim1 lo
  unsigned w3 = (tensor_d1 >> 16) | ((tile_d0 & 0xffffu) << 16);   // dim1 hi, tile_dim0
  unsigned w4 = (tile_d1 & 0xffffu);                               // tile_dim1, tile_dim2=0
  unsigned w5 = (unsigned)(stride0 & 0xffffffffu);                 // dim0_stride lo
  unsigned w6 = (unsigned)((stride0 >> 32) & 0xffffu);             // dim0_stride hi
  unsigned w7 = 0u;
  v8i_t g1;
  g1[0]=(int)w0; g1[1]=(int)w1; g1[2]=(int)w2; g1[3]=(int)w3;
  g1[4]=(int)w4; g1[5]=(int)w5; g1[6]=(int)w6; g1[7]=(int)w7;
  v4i_t z4; z4[0]=0; z4[1]=0; z4[2]=0; z4[3]=0;   // groups 2/3 unused (2D)
  v8i_t z8; z8[0]=0; z8[1]=0; z8[2]=0; z8[3]=0; z8[4]=0; z8[5]=0; z8[6]=0; z8[7]=0;
  __builtin_amdgcn_tensor_load_to_lds(g0, g1, z4, z4, z8, 0);
}
#endif

__device__ __forceinline__ void wait_tensor0() {
#if __has_builtin(__builtin_amdgcn_s_wait_tensorcnt)
  __builtin_amdgcn_s_wait_tensorcnt(0);
#else
  asm volatile("s_wait_tensorcnt 0x0" ::: "memory");
#endif
}

struct Bf16Pair { __bf16 h; __bf16 l; };
__device__ __forceinline__ Bf16Pair split_bf16(float f) {
  Bf16Pair p;
  p.h = (__bf16)f;
  p.l = (__bf16)(f - (float)p.h);
  return p;
}

// ---- Pre-pass: split W[64,4096] f32 into bf16 hi/lo planes in d_ws ----
__global__ __launch_bounds__(256) void w_split_kernel(const float* __restrict__ W,
                                                      unsigned short* __restrict__ wh,
                                                      unsigned short* __restrict__ wl) {
  const int i = blockIdx.x * 256 + threadIdx.x;   // grid covers 64*4096 elements
  const Bf16Pair p = split_bf16(W[i]);
  wh[i] = __builtin_bit_cast(unsigned short, p.h);
  wl[i] = __builtin_bit_cast(unsigned short, p.l);
}

template <bool WPRE>
__global__ __launch_bounds__(256)
void moe_router_kernel(const float* __restrict__ x, const float* __restrict__ W,
                       const unsigned short* __restrict__ wh,
                       const unsigned short* __restrict__ wl,
                       float* __restrict__ logits, float* __restrict__ rweights,
                       int* __restrict__ ridx) {
  extern __shared__ float smem[];
  const unsigned tid  = threadIdx.x;
  const unsigned wave = tid >> 5;
  const unsigned lane = tid & 31;
  const unsigned nl   = lane & 15;   // N-column / M-row within fragment
  const unsigned half = lane >> 4;   // K-half selector
  const int block = blockIdx.x;

  // LDS layout (dwords)
  const unsigned xoff[2]  = {0u, (unsigned)XBUF};
  const unsigned whoff[2] = {2u * XBUF, 2u * XBUF + WHBUF};              // WPRE: bf16 hi
  const unsigned wloff[2] = {2u * XBUF + 2 * WHBUF, 2u * XBUF + 3 * WHBUF}; // WPRE: bf16 lo
  const unsigned wfoff[2] = {2u * XBUF, 2u * XBUF + WBUF};               // fallback: f32 W

  v8f acc0 = {}, acc1 = {}, acc2 = {}, acc3 = {};

  auto load_chunk = [&](int k0, int buf) {
#if USE_TDM
    if (tid < 32) {  // wave 0 issues DMA; EXEC ignored by TDM
      const float* xg = x + (size_t)block * M_BLOCK * HIDDEN + k0;
      tdm_issue_2d(xoff[buf] * 4u, xg, KC, M_BLOCK, HIDDEN, KC, M_BLOCK, 2u, 4u);
      if constexpr (WPRE) {
        tdm_issue_2d(whoff[buf] * 4u, wh + k0, KC, EXPERTS, HIDDEN, KC, EXPERTS, 1u, 3u);
        tdm_issue_2d(wloff[buf] * 4u, wl + k0, KC, EXPERTS, HIDDEN, KC, EXPERTS, 1u, 3u);
      } else {
        tdm_issue_2d(wfoff[buf] * 4u, W + k0, KC, EXPERTS, HIDDEN, KC, EXPERTS, 2u, 4u);
      }
    }
#else
    const float* xg = x + (size_t)block * M_BLOCK * HIDDEN + k0;
    for (int i = tid; i < M_BLOCK * KC; i += 256) {
      int r = i >> 5, c = i & 31;
      smem[xoff[buf] + r * XSTRIDE + c] = xg[(size_t)r * HIDDEN + c];
    }
    if constexpr (WPRE) {
      unsigned short* sh = (unsigned short*)smem;
      for (int i = tid; i < EXPERTS * KC; i += 256) {
        int r = i >> 5, c = i & 31;
        sh[whoff[buf] * 2 + r * 34 + c] = wh[(size_t)r * HIDDEN + k0 + c];
        sh[wloff[buf] * 2 + r * 34 + c] = wl[(size_t)r * HIDDEN + k0 + c];
      }
    } else {
      for (int i = tid; i < EXPERTS * KC; i += 256) {
        int r = i >> 5, c = i & 31;
        smem[wfoff[buf] + r * XSTRIDE + c] = W[(size_t)r * HIDDEN + k0 + c];
      }
    }
#endif
  };

  // Prime buffer 0
  load_chunk(0, 0);
#if USE_TDM
  if (tid < 32) wait_tensor0();
#endif
  __syncthreads();

  const int nchunks = HIDDEN / KC;  // 128
#pragma unroll 1
  for (int c = 0; c < nchunks; ++c) {
    const int cur = c & 1, nxt = cur ^ 1;
    if (c + 1 < nchunks) load_chunk((c + 1) * KC, nxt);

    // ---- A fragment: 16 tokens x 32 K, fp32 -> bf16 hi/lo split ----
    const float* xr = smem + xoff[cur] + (wave * 16 + nl) * XSTRIDE + half * 16;
    v16bf ah, al;
#pragma unroll
    for (int i = 0; i < 8; ++i) {
      Bf16Pair p = split_bf16(xr[i]);
      ah[i] = p.h; al[i] = p.l;
    }
#pragma unroll
    for (int i = 0; i < 8; ++i) {
      Bf16Pair p = split_bf16(xr[8 + i]);
      ah[8 + i] = p.h; al[8 + i] = p.l;
    }

    // ---- B fragments: 4 expert tiles of 32K x 16N ----
#pragma unroll
    for (int t = 0; t < 4; ++t) {
      v16bf bh, bl;
      if constexpr (WPRE) {
        // pre-split bf16 planes: pure LDS loads, zero conversion VALU
        union Frag { unsigned u[8]; v16bf v; };
        const unsigned* su = (const unsigned*)smem;
        const unsigned rb = (t * 16 + nl) * WHSTRIDE + half * 8;
        Frag fh, fl;
#pragma unroll
        for (int j = 0; j < 8; ++j) { fh.u[j] = su[whoff[cur] + rb + j]; }
#pragma unroll
        for (int j = 0; j < 8; ++j) { fl.u[j] = su[wloff[cur] + rb + j]; }
        bh = fh.v; bl = fl.v;
      } else {
        const float* wp = smem + wfoff[cur] + (t * 16 + nl) * XSTRIDE + half * 16;
#pragma unroll
        for (int i = 0; i < 16; ++i) {
          Bf16Pair p = split_bf16(wp[i]);
          bh[i] = p.h; bl[i] = p.l;
        }
      }
      v8f& acc = (t == 0) ? acc0 : (t == 1) ? acc1 : (t == 2) ? acc2 : acc3;
      // fp32-accurate: hi*hi + hi*lo + lo*hi (lo*lo ~ 2^-32, dropped)
      acc = __builtin_amdgcn_wmma_f32_16x16x32_bf16(false, ah, false, bh, (short)0, acc, false, false);
      acc = __builtin_amdgcn_wmma_f32_16x16x32_bf16(false, ah, false, bl, (short)0, acc, false, false);
      acc = __builtin_amdgcn_wmma_f32_16x16x32_bf16(false, al, false, bh, (short)0, acc, false, false);
    }

#if USE_TDM
    if (tid < 32) wait_tensor0();
#endif
    __syncthreads();
  }

  // ---- Epilogue: write logits + fused top-2 softmax routing ----
  __syncthreads();                      // all chunk reads done; safe to reuse LDS
  float* scratch = smem + wave * 1060;  // per-wave 16x65 padded logit tile
  const int tokBase = block * M_BLOCK + wave * 16;
#pragma unroll
  for (int t = 0; t < 4; ++t) {
    v8f acc = (t == 0) ? acc0 : (t == 1) ? acc1 : (t == 2) ? acc2 : acc3;
#pragma unroll
    for (int j = 0; j < 8; ++j) {
      const int m = (int)half * 8 + j;     // token row in C fragment
      const int e = t * 16 + (int)nl;      // expert column
      const float v = acc[j];
      logits[(size_t)(tokBase + m) * EXPERTS + e] = v;
      scratch[m * 65 + e] = v;
    }
  }
  asm volatile("s_wait_dscnt 0x0" ::: "memory");  // same-wave LDS RAW fence

  if (lane < 16) {
    const float* row = scratch + lane * 65;
    float m1 = -1e30f, m2 = -1e30f;
    int i1 = 0, i2 = 0;
#pragma unroll 8
    for (int e = 0; e < EXPERTS; ++e) {
      const float v = row[e];
      if (v > m1)      { m2 = m1; i2 = i1; m1 = v; i1 = e; }
      else if (v > m2) { m2 = v; i2 = e; }
    }
    // normalized top-2 softmax weights: 1/(1+e), e/(1+e) with e = exp(l2-l1)
    const float e2  = __expf(m2 - m1);
    const float inv = 1.0f / (1.0f + e2);
    const int tok = tokBase + (int)lane;
    rweights[tok * 2 + 0] = inv;
    rweights[tok * 2 + 1] = e2 * inv;
    ridx[tok * 2 + 0] = i1;
    ridx[tok * 2 + 1] = i2;
  }
}

extern "C" void kernel_launch(void* const* d_in, const int* in_sizes, int n_in,
                              void* d_out, int out_size, void* d_ws, size_t ws_size,
                              hipStream_t stream) {
  const float* x = (const float*)d_in[0];   // [4, 8192, 4096] f32
  const float* W = (const float*)d_in[1];   // [64, 4096] f32
  float* out    = (float*)d_out;
  float* logits = out;                                   // [32768, 64]
  float* rw     = out + (size_t)T_TOKENS * EXPERTS;      // [32768, 2]
  int*   ridx   = (int*)(rw + (size_t)T_TOKENS * 2);     // [32768, 2]

  const size_t nW      = (size_t)EXPERTS * HIDDEN;       // 262,144 elements
  const size_t wsNeed  = nW * 2 * sizeof(unsigned short); // 1 MB: bf16 hi + lo planes
  const size_t smemBytes = (size_t)SMEM_DW * sizeof(float); // 51,584 B

  dim3 grid(T_TOKENS / M_BLOCK);  // 256 workgroups
  dim3 blockDim(256);             // 8 wave32 waves

  if (ws_size >= wsNeed) {
    unsigned short* wh = (unsigned short*)d_ws;
    unsigned short* wl = wh + nW;
    w_split_kernel<<<dim3(nW / 256), blockDim, 0, stream>>>(W, wh, wl);
    moe_router_kernel<true><<<grid, blockDim, smemBytes, stream>>>(x, W, wh, wl,
                                                                   logits, rw, ridx);
  } else {
    moe_router_kernel<false><<<grid, blockDim, smemBytes, stream>>>(x, W, nullptr, nullptr,
                                                                    logits, rw, ridx);
  }
}
